// MoEFFN_41936060678259
// MI455X (gfx1250) — compile-verified, matrix operand order checked by reading, not run on hardware
//
#include <hip/hip_runtime.h>

// ---------------------------------------------------------------------------
// MoE FFN for MI455X (gfx1250, wave32). bf16 WMMA 16x16x32 with f32 accum.
// LDS strides padded for 64-bank conflict avoidance.
// ---------------------------------------------------------------------------

typedef __bf16 bf16_t;
typedef bf16_t v16bf __attribute__((ext_vector_type(16)));
typedef bf16_t v8bf  __attribute__((ext_vector_type(8)));
typedef bf16_t v2bf  __attribute__((ext_vector_type(2)));
typedef float  v8f   __attribute__((ext_vector_type(8)));
typedef float  v4f   __attribute__((ext_vector_type(4)));

#define N_TOK   2048
#define DMODEL  2048
#define NEXP    64
#define TOPK    4
#define FEXP    512
#define NSHARED 2
#define HSHARED 2048

// LDS strides (elements). Chosen so row/col dword strides are != 0 mod 64
// while preserving 16B alignment for ds_load_b128 fragment reads.
#define LDA  2056   // Ash row stride  (dword stride 1028 = 4 mod 64)
#define LDHR 520    // routed Hs row stride (dword stride 260 = 4 mod 64)
#define LDB  40     // BT column pitch (80B; read stride 20 dw, gcd(20,64)=4)

// ---- bf16 convert/pack: prefer single-op v_cvt_pk_bf16_f32 ------------------
#if __has_builtin(__builtin_amdgcn_cvt_pk_bf16_f32)
__device__ inline unsigned pack2(float a, float b) {
    v2bf p = __builtin_amdgcn_cvt_pk_bf16_f32(a, b);
    return __builtin_bit_cast(unsigned, p);
}
__device__ inline bf16_t f2bf(float f) {
    v2bf p = __builtin_amdgcn_cvt_pk_bf16_f32(f, f);
    return p[0];
}
#else
__device__ inline unsigned pack2(float a, float b) {
    v2bf p;
    p[0] = (bf16_t)a;
    p[1] = (bf16_t)b;
    return __builtin_bit_cast(unsigned, p);
}
__device__ inline bf16_t f2bf(float f) { return (bf16_t)f; }
#endif

__device__ inline v8f zero8() {
    v8f z = {0.f, 0.f, 0.f, 0.f, 0.f, 0.f, 0.f, 0.f};
    return z;
}

// Fast SiLU: one v_rcp_f32 TRANS op instead of the IEEE divide expansion.
__device__ inline float silu_fast(float g) {
    return g * __builtin_amdgcn_rcpf(1.f + __expf(-g));
}

// ---- WMMA wrapper -----------------------------------------------------------
__device__ inline v8f wmma_bf16(v16bf a, v16bf b, v8f c) {
    return __builtin_amdgcn_wmma_f32_16x16x32_bf16(
        /*neg_a=*/false, a, /*neg_b=*/false, b,
        /*c_mod=*/(short)0, c, /*reuse_a=*/false, /*reuse_b=*/false);
}

// Load a 16-elem bf16 fragment for one lane from LDS (row-major, stride lda).
// 16-bit A/B layout (ISA 7.12.2): lane L covers row (L&15); its 16 elems are
// two runs of 8 contiguous K: [k0+kb .. +7] and [k0+16+kb .. +7], kb=(L>>4)*8.
__device__ inline v16bf ld_frag(const bf16_t* base, int row, int lda, int k0, int lane) {
    const int kb = (lane >> 4) * 8;
    const bf16_t* p = base + row * lda + k0 + kb;
    v8bf lo = *(const v8bf*)(p);
    v8bf hi = *(const v8bf*)(p + 16);
    v16bf r;
#pragma unroll
    for (int i = 0; i < 8; ++i) { r[i] = lo[i]; r[8 + i] = hi[i]; }
    return r;
}

// Cooperatively stage a 32(K) x 512(cols) fp32 tile of row-major W[ldw] into
// transposed bf16 LDS BT[col][LDB]. Adjacent lanes take adjacent 4-col groups:
// global float4 reads coalesce (256B/row/wave); LDS b32 writes have lane
// dword-stride 4*20=80 = 16 mod 64 (4-way, vs 16-way unpadded). Fragment
// reads (col stride 20 dwords) are conflict-free. Prefetch next K rows.
__device__ inline void stage_bt(bf16_t* BT, const float* W, int ldw,
                                int k0, int cb, int tid) {
    const int kr = (tid >> 4) * 2;     // 0,2,...,30
    const int f0 = (tid & 15) * 4;     // adjacent lanes -> adjacent 4 cols
    const float* r0 = W + (size_t)(k0 + kr) * ldw + cb;
    const float* r1 = r0 + ldw;
    __builtin_prefetch(r0 + (size_t)32 * ldw, 0, 0);  // speculative, next K tile
#pragma unroll
    for (int j = 0; j < 8; ++j) {
        int f = f0 + j * 64;
        v4f a = *(const v4f*)(r0 + f);
        v4f b = *(const v4f*)(r1 + f);
#pragma unroll
        for (int c = 0; c < 4; ++c) {
            *(unsigned*)(BT + (size_t)(f + c) * LDB + kr) = pack2(a[c], b[c]);
        }
    }
}

// Gather one 16 x DMODEL fp32 row tile into bf16 LDS (row stride LDA).
// Each lane packs 8 bf16 into a uint4 and issues one ds_store_b128;
// lane dword-stride 4 -> all 64 banks covered, conflict-free.
__device__ inline void gather_tile(bf16_t* Ash, const float* src, int row, int tid) {
    const int c8 = (tid & 15) * 8;
    bf16_t* dst = Ash + row * LDA;
#pragma unroll
    for (int j = 0; j < 16; ++j) {
        int cb = c8 + j * 128;
        v4f a = *(const v4f*)(src + cb);
        v4f b = *(const v4f*)(src + cb + 4);
        uint4 p;
        p.x = pack2(a[0], a[1]);
        p.y = pack2(a[2], a[3]);
        p.z = pack2(b[0], b[1]);
        p.w = pack2(b[2], b[3]);
        *(uint4*)(dst + cb) = p;
    }
}

// ---------------------------------------------------------------------------
// Kernel 0: zero output + expert counters
// ---------------------------------------------------------------------------
__global__ void moe_zero_kernel(float* __restrict__ out, int* __restrict__ cnt) {
    int i = blockIdx.x * blockDim.x + threadIdx.x;
    if (i < (N_TOK * DMODEL) / 4) ((v4f*)out)[i] = zero8().lo;  // 4-wide zero
    if (i < NEXP) cnt[i] = 0;
}

// ---------------------------------------------------------------------------
// Kernel 1: router — logits, softmax, top-4 (on logits+bias), renormalized
// combine weights, per-expert counts. One block (64 threads) per token.
// ---------------------------------------------------------------------------
__global__ __launch_bounds__(64)
void moe_router_kernel(const float* __restrict__ x,
                       const float* __restrict__ rw,
                       const float* __restrict__ rbias,
                       int* __restrict__ topk_idx,
                       float* __restrict__ topk_w,
                       int* __restrict__ cnt) {
    __shared__ float xrow[DMODEL];
    __shared__ float lg[NEXP];
    const int n = blockIdx.x;
    const int t = threadIdx.x;

#pragma unroll
    for (int j = 0; j < 8; ++j) {
        int off = j * 256 + t * 4;
        *(v4f*)(xrow + off) = *(const v4f*)(x + (size_t)n * DMODEL + off);
    }
    __syncthreads();

    float acc = 0.f;
    for (int d = 0; d < DMODEL; ++d) acc += xrow[d] * rw[(size_t)d * NEXP + t];
    lg[t] = acc;
    __syncthreads();

    if (t == 0) {
        float m = -1e30f;
        for (int e = 0; e < NEXP; ++e) m = fmaxf(m, lg[e]);
        float ssum = 0.f;
        for (int e = 0; e < NEXP; ++e) ssum += __expf(lg[e] - m);
        unsigned long long taken = 0ull;
        int sel[TOPK]; float sw[TOPK]; float wsum = 0.f;
        for (int k = 0; k < TOPK; ++k) {
            int best = 0; float bv = -1e30f;
            for (int e = 0; e < NEXP; ++e) {
                if (taken & (1ull << e)) continue;
                float v = lg[e] + rbias[e];
                if (v > bv) { bv = v; best = e; }
            }
            taken |= (1ull << best);
            sel[k] = best;
            sw[k] = __expf(lg[best] - m) / ssum;
            wsum += sw[k];
        }
        for (int k = 0; k < TOPK; ++k) {
            topk_idx[n * TOPK + k] = sel[k];
            topk_w[n * TOPK + k] = sw[k] / wsum;
            atomicAdd(&cnt[sel[k]], 1);
        }
    }
}

// ---------------------------------------------------------------------------
// Kernel 2: exclusive scan of expert counts (tiny, E=64)
// ---------------------------------------------------------------------------
__global__ void moe_scan_kernel(const int* __restrict__ cnt,
                                int* __restrict__ offs,
                                int* __restrict__ cursor) {
    if (threadIdx.x == 0 && blockIdx.x == 0) {
        int r = 0;
        for (int e = 0; e < NEXP; ++e) { offs[e] = r; cursor[e] = r; r += cnt[e]; }
        offs[NEXP] = r;
    }
}

// ---------------------------------------------------------------------------
// Kernel 3: scatter assignments into compact per-expert token lists
// ---------------------------------------------------------------------------
__global__ void moe_scatter_kernel(const int* __restrict__ topk_idx,
                                   const float* __restrict__ topk_w,
                                   int* __restrict__ cursor,
                                   int* __restrict__ list_tok,
                                   float* __restrict__ list_w) {
    int i = blockIdx.x * blockDim.x + threadIdx.x;
    if (i < N_TOK * TOPK) {
        int e = topk_idx[i];
        int pos = atomicAdd(&cursor[e], 1);
        list_tok[pos] = i / TOPK;
        list_w[pos]   = topk_w[i];
    }
}

// ---------------------------------------------------------------------------
// Kernel 4: routed experts. Block = 256 threads (8 waves). grid = (E, 8).
// Each block: expert e, strided loop over 16-token tiles of its list.
// LDS: Ash 64.2KB + Hs 16.3KB + 2x40KB staging = ~161KB.
// ---------------------------------------------------------------------------
__global__ __launch_bounds__(256)
void moe_routed_kernel(const float* __restrict__ x,
                       const float* __restrict__ wg,
                       const float* __restrict__ wu,
                       const float* __restrict__ wd,
                       const int* __restrict__ offs,
                       const int* __restrict__ list_tok,
                       const float* __restrict__ list_w,
                       float* __restrict__ out) {
    __shared__ bf16_t Ash[16 * LDA];      // activations, bf16
    __shared__ bf16_t Hs[16 * LDHR];      // silu(g)*u hidden, bf16
    __shared__ bf16_t BT0[FEXP * LDB];    // transposed weight tile (gate / down)
    __shared__ bf16_t BT1[FEXP * LDB];    // transposed weight tile (up)
    __shared__ int    toks[16];
    __shared__ float  tw[16];

    const int e    = blockIdx.x;
    const int tid  = threadIdx.x;
    const int wave = tid >> 5;
    const int lane = tid & 31;
    const int base = offs[e];
    const int cnte = offs[e + 1] - base;
    const float* Wg = wg + (size_t)e * DMODEL * FEXP;
    const float* Wu = wu + (size_t)e * DMODEL * FEXP;
    const float* Wd = wd + (size_t)e * FEXP * DMODEL;

    for (int tile = blockIdx.y; tile * 16 < cnte; tile += gridDim.y) {
        if (tid < 16) {
            int i = tile * 16 + tid;
            toks[tid] = (i < cnte) ? list_tok[base + i] : 0;
            tw[tid]   = (i < cnte) ? list_w[base + i] : 0.f;
        }
        __syncthreads();

        gather_tile(Ash, x + (size_t)toks[tid >> 4] * DMODEL, tid >> 4, tid);
        __syncthreads();

        // --- gate + up: [16 x D] @ [D x F], F=512 split 64 cols per wave ---
        v8f accG[4], accU[4];
#pragma unroll
        for (int t4 = 0; t4 < 4; ++t4) { accG[t4] = zero8(); accU[t4] = zero8(); }

        for (int k0 = 0; k0 < DMODEL; k0 += 32) {
            stage_bt(BT0, Wg, FEXP, k0, 0, tid);
            stage_bt(BT1, Wu, FEXP, k0, 0, tid);
            __syncthreads();
            v16bf a = ld_frag(Ash, lane & 15, LDA, k0, lane);
#pragma unroll
            for (int t4 = 0; t4 < 4; ++t4) {
                int col = wave * 64 + t4 * 16 + (lane & 15);
                v16bf bg = ld_frag(BT0, col, LDB, 0, lane);
                v16bf bu = ld_frag(BT1, col, LDB, 0, lane);
                accG[t4] = wmma_bf16(a, bg, accG[t4]);
                accU[t4] = wmma_bf16(a, bu, accU[t4]);
            }
            __syncthreads();
        }

        // SwiGLU on accumulators -> Hs (C/D layout: M = r + 8*(lane>=16), N = lane&15)
        {
            const int m0 = (lane >> 4) * 8;
            const int nn = lane & 15;
#pragma unroll
            for (int t4 = 0; t4 < 4; ++t4) {
                int f = wave * 64 + t4 * 16 + nn;
#pragma unroll
                for (int r = 0; r < 8; ++r) {
                    Hs[(m0 + r) * LDHR + f] = f2bf(silu_fast(accG[t4][r]) * accU[t4][r]);
                }
            }
        }
        __syncthreads();

        // --- down: [16 x F] @ [F x D], D=2048 in four 512-col chunks ---
        for (int nc = 0; nc < 4; ++nc) {
            const int db = nc * 512;
            v8f acc[4];
#pragma unroll
            for (int t4 = 0; t4 < 4; ++t4) acc[t4] = zero8();
            for (int k0 = 0; k0 < FEXP; k0 += 32) {
                stage_bt(BT0, Wd, DMODEL, k0, db, tid);
                __syncthreads();
                v16bf a = ld_frag(Hs, lane & 15, LDHR, k0, lane);
#pragma unroll
                for (int t4 = 0; t4 < 4; ++t4) {
                    int col = wave * 64 + t4 * 16 + (lane & 15);
                    v16bf b = ld_frag(BT0, col, LDB, 0, lane);
                    acc[t4] = wmma_bf16(a, b, acc[t4]);
                }
                __syncthreads();
            }
            const int m0 = (lane >> 4) * 8;
            const int nn = lane & 15;
#pragma unroll
            for (int t4 = 0; t4 < 4; ++t4) {
                int d = db + wave * 64 + t4 * 16 + nn;
#pragma unroll
                for (int r = 0; r < 8; ++r) {
                    int m = m0 + r;
                    atomicAdd(&out[(size_t)toks[m] * DMODEL + d], acc[t4][r] * tw[m]);
                }
            }
        }
        __syncthreads();
    }
}

// ---------------------------------------------------------------------------
// Kernel 5: shared experts (dense). grid = (N/16, S). Block = 256 threads.
// LDS: Ash 64.2KB + Hs 64.2KB + 2x40KB staging = ~209KB (<= 320KB WGP LDS).
// ---------------------------------------------------------------------------
__global__ __launch_bounds__(256)
void moe_shared_kernel(const float* __restrict__ x,
                       const float* __restrict__ swg,
                       const float* __restrict__ swu,
                       const float* __restrict__ swd,
                       float* __restrict__ out) {
    __shared__ bf16_t Ash[16 * LDA];
    __shared__ bf16_t Hs[16 * LDA];
    __shared__ bf16_t BT0[512 * LDB];
    __shared__ bf16_t BT1[512 * LDB];

    const int n0   = blockIdx.x * 16;
    const int s    = blockIdx.y;
    const int tid  = threadIdx.x;
    const int wave = tid >> 5;
    const int lane = tid & 31;
    const float* Wg = swg + (size_t)s * DMODEL * HSHARED;
    const float* Wu = swu + (size_t)s * DMODEL * HSHARED;
    const float* Wd = swd + (size_t)s * HSHARED * DMODEL;

    gather_tile(Ash, x + (size_t)(n0 + (tid >> 4)) * DMODEL, tid >> 4, tid);
    __syncthreads();

    // --- gate + up over HS in four 512-wide chunks ---
    for (int fc = 0; fc < 4; ++fc) {
        const int fb = fc * 512;
        v8f accG[4], accU[4];
#pragma unroll
        for (int t4 = 0; t4 < 4; ++t4) { accG[t4] = zero8(); accU[t4] = zero8(); }
        for (int k0 = 0; k0 < DMODEL; k0 += 32) {
            stage_bt(BT0, Wg, HSHARED, k0, fb, tid);
            stage_bt(BT1, Wu, HSHARED, k0, fb, tid);
            __syncthreads();
            v16bf a = ld_frag(Ash, lane & 15, LDA, k0, lane);
#pragma unroll
            for (int t4 = 0; t4 < 4; ++t4) {
                int col = wave * 64 + t4 * 16 + (lane & 15);
                v16bf bg = ld_frag(BT0, col, LDB, 0, lane);
                v16bf bu = ld_frag(BT1, col, LDB, 0, lane);
                accG[t4] = wmma_bf16(a, bg, accG[t4]);
                accU[t4] = wmma_bf16(a, bu, accU[t4]);
            }
            __syncthreads();
        }
        const int m0 = (lane >> 4) * 8;
        const int nn = lane & 15;
#pragma unroll
        for (int t4 = 0; t4 < 4; ++t4) {
            int h = fb + wave * 64 + t4 * 16 + nn;
#pragma unroll
            for (int r = 0; r < 8; ++r) {
                Hs[(m0 + r) * LDA + h] = f2bf(silu_fast(accG[t4][r]) * accU[t4][r]);
            }
        }
        __syncthreads();
    }

    // --- down: [16 x HS] @ [HS x D] ---
    for (int nc = 0; nc < 4; ++nc) {
        const int db = nc * 512;
        v8f acc[4];
#pragma unroll
        for (int t4 = 0; t4 < 4; ++t4) acc[t4] = zero8();
        for (int k0 = 0; k0 < HSHARED; k0 += 32) {
            stage_bt(BT0, Wd, DMODEL, k0, db, tid);
            __syncthreads();
            v16bf a = ld_frag(Hs, lane & 15, LDA, k0, lane);
#pragma unroll
            for (int t4 = 0; t4 < 4; ++t4) {
                int col = wave * 64 + t4 * 16 + (lane & 15);
                v16bf b = ld_frag(BT0, col, LDB, 0, lane);
                acc[t4] = wmma_bf16(a, b, acc[t4]);
            }
            __syncthreads();
        }
        const int m0 = (lane >> 4) * 8;
        const int nn = lane & 15;
#pragma unroll
        for (int t4 = 0; t4 < 4; ++t4) {
            int d = db + wave * 64 + t4 * 16 + nn;
#pragma unroll
            for (int r = 0; r < 8; ++r) {
                atomicAdd(&out[(size_t)(n0 + m0 + r) * DMODEL + d], acc[t4][r]);
            }
        }
    }
}

// ---------------------------------------------------------------------------
extern "C" void kernel_launch(void* const* d_in, const int* in_sizes, int n_in,
                              void* d_out, int out_size, void* d_ws, size_t ws_size,
                              hipStream_t stream) {
    (void)in_sizes; (void)n_in; (void)out_size; (void)ws_size;
    const float* x   = (const float*)d_in[0];
    const float* rw  = (const float*)d_in[1];
    const float* rb  = (const float*)d_in[2];
    const float* wg  = (const float*)d_in[3];
    const float* wu  = (const float*)d_in[4];
    const float* wd  = (const float*)d_in[5];
    const float* swg = (const float*)d_in[6];
    const float* swu = (const float*)d_in[7];
    const float* swd = (const float*)d_in[8];
    float* out = (float*)d_out;

    // Workspace layout (~129KB)
    int*   topk_idx = (int*)d_ws;                       // N*K
    float* topk_w   = (float*)(topk_idx + N_TOK * TOPK);
    int*   cnt      = (int*)(topk_w + N_TOK * TOPK);    // E
    int*   offs     = cnt + NEXP;                       // E+1
    int*   cursor   = offs + NEXP + 1;                  // E
    int*   list_tok = cursor + NEXP;                    // N*K
    float* list_w   = (float*)(list_tok + N_TOK * TOPK);

    moe_zero_kernel<<<(N_TOK * DMODEL / 4 + 255) / 256, 256, 0, stream>>>(out, cnt);
    moe_router_kernel<<<N_TOK, 64, 0, stream>>>(x, rw, rb, topk_idx, topk_w, cnt);
    moe_scan_kernel<<<1, 32, 0, stream>>>(cnt, offs, cursor);
    moe_scatter_kernel<<<(N_TOK * TOPK + 255) / 256, 256, 0, stream>>>(
        topk_idx, topk_w, cursor, list_tok, list_w);
    moe_routed_kernel<<<dim3(NEXP, 8), 256, 0, stream>>>(
        x, wg, wu, wd, offs, list_tok, list_w, out);
    moe_shared_kernel<<<dim3(N_TOK / 16, NSHARED), 256, 0, stream>>>(
        x, swg, swu, swd, out);
}
